// OverlapWindowMHA2d_44358422233724
// MI455X (gfx1250) — compile-verified, matrix operand order checked by reading, not run on hardware
//
#include <hip/hip_runtime.h>
#include <hip/hip_bf16.h>

// ---------------------------------------------------------------------------
// OverlapWindowMHA2d for MI455X (gfx1250, wave32, WMMA, async-to-LDS)
//   B=4, C=384, H=W=128, heads=6, d=64, window 8x8, pad 4
// Stage 1: qkv(bf16) = w_qkv @ x + b_qkv    (M=1152,K=384,N=16384 per batch)
// Stage 2: window MHA (aligned -> x1 bf16, shifted -> x2 bf16)
// Stage 3: out(f32) = w_head @ (x1+x2) + b_head
// v_wmma_f32_16x16x32_bf16 everywhere; async global->LDS for V staging;
// all LDS strides padded for the 64-bank x 4B LDS (conflict-free b128 frags).
// ---------------------------------------------------------------------------

typedef __attribute__((ext_vector_type(16))) __bf16 v16bf;
typedef __attribute__((ext_vector_type(8)))  float  v8f;

#define HWPIX 16384   // 128*128
#define IMG   128
#define KDIM  384     // K for both GEMMs
#define SBSTR 392     // GEMM LDS panel row stride (bf16): 196 dwords == 4 mod 64
#define TSTR  72      // attention tile row stride (bf16): 36 dwords, 16 distinct banks
#define SSTR  65      // score row stride (f32): 65 dwords == 1 mod 64

// 16x32 bf16 A/B fragment (ISA 7.12.2): lanes 0-15: row=lane, K 0..7 + 16..23;
// lanes 16-31: row=lane-16, K 8..15 + 24..31.  rowp at (row, ktile), kh=(lane>>4)*8.
static __device__ __forceinline__ v16bf load_frag16(const __bf16* rowp, int kh) {
  v16bf v;
  ((uint4*)&v)[0] = *(const uint4*)(rowp + kh);
  ((uint4*)&v)[1] = *(const uint4*)(rowp + kh + 16);
  return v;
}

// Async DMA: 16 contiguous global bytes -> 16 contiguous LDS bytes (per lane).
static __device__ __forceinline__ void async_copy_b128(unsigned lds_off, const void* gptr) {
  asm volatile("global_load_async_to_lds_b128 %0, %1, off"
               :: "v"(lds_off), "v"(gptr) : "memory");
}
static __device__ __forceinline__ void wait_async0() {
  asm volatile("s_wait_asynccnt 0x0" ::: "memory");
}

// ---------------------------------------------------------------------------
// Stages 1 & 3: OUT[b,m,n] = sum_k W[m,k] * ACT[b,k,n] + bias[m]
//   ACT_SUM_BF16=0: act0 is f32;  =1: act = bf16 act0 + bf16 act1
//   OUT_BF16=1: bf16 output, else f32.
// grid = (N/64, 1, B), block = 256 (8 wave32). Full-K activation panel in LDS
// (64 x 392 bf16 ~= 49KB), staged once; loop over all M tiles; A fragments are
// loaded straight from global (weights are L2-resident) and converted.
// ---------------------------------------------------------------------------
template <bool ACT_SUM_BF16, bool OUT_BF16>
__global__ __launch_bounds__(256)
void wmma_gemm_full(const float* __restrict__ Wm,     // M x 384 row-major f32
                    const void*  __restrict__ act0v,  // B x 384 x N
                    const void*  __restrict__ act1v,  // B x 384 x N (or null)
                    const float* __restrict__ bias,   // M
                    void*        __restrict__ outv,   // B x M x N
                    int M, int N)
{
  __shared__ __attribute__((aligned(16))) __bf16 sB[64 * SBSTR];  // [n][k], padded

  const int t    = threadIdx.x;
  const int lane = t & 31;
  const int wave = t >> 5;
  const int b    = blockIdx.z;
  const int n0   = blockIdx.x * 64;

  // ---- stage full-K activation panel (transpose to [n][k]) ----
  {
    const float*  a0f = (const float*)act0v  + (size_t)b * KDIM * N;
    const __bf16* a0b = (const __bf16*)act0v + (size_t)b * KDIM * N;
    const __bf16* a1b = (const __bf16*)act1v + (size_t)b * KDIM * N;
#pragma unroll 4
    for (int i = 0; i < (64 * KDIM) / 256; ++i) {
      int idx = i * 256 + t;
      int k = idx >> 6;          // 0..383
      int n = idx & 63;          // 0..63
      size_t g = (size_t)k * N + n0 + n;
      float v;
      if (ACT_SUM_BF16) v = (float)a0b[g] + (float)a1b[g];
      else              v = a0f[g];
      sB[n * SBSTR + k] = (__bf16)v;
    }
  }
  __syncthreads();   // only barrier in the kernel

  const int frow = lane & 15;
  const int fkh  = (lane >> 4) * 8;
  const int mrow = (lane >> 4) * 8;
  const int ncol = lane & 15;

  float*  outf = (float*)outv   + (size_t)b * M * N;
  __bf16* outb = (__bf16*)outv  + (size_t)b * M * N;

  for (int m0 = 0; m0 < M; m0 += 128) {
    v8f acc[4] = {{}, {}, {}, {}};
    const float* arow = Wm + (size_t)(m0 + wave * 16 + frow) * KDIM + fkh;

#pragma unroll 4
    for (int k0 = 0; k0 < KDIM; k0 += 32) {
      if (k0 + 32 < KDIM)
        __builtin_prefetch(arow + k0 + 32, 0, 3);   // global_prefetch_b8
      // A fragment: 16 f32 from global (L2-hot weights), convert to bf16
      const float* ap = arow + k0;
      float4 fa = *(const float4*)(ap);
      float4 fb = *(const float4*)(ap + 4);
      float4 fc = *(const float4*)(ap + 16);
      float4 fd = *(const float4*)(ap + 20);
      v16bf a;
      a[ 0] = (__bf16)fa.x; a[ 1] = (__bf16)fa.y; a[ 2] = (__bf16)fa.z; a[ 3] = (__bf16)fa.w;
      a[ 4] = (__bf16)fb.x; a[ 5] = (__bf16)fb.y; a[ 6] = (__bf16)fb.z; a[ 7] = (__bf16)fb.w;
      a[ 8] = (__bf16)fc.x; a[ 9] = (__bf16)fc.y; a[10] = (__bf16)fc.z; a[11] = (__bf16)fc.w;
      a[12] = (__bf16)fd.x; a[13] = (__bf16)fd.y; a[14] = (__bf16)fd.z; a[15] = (__bf16)fd.w;
#pragma unroll
      for (int j = 0; j < 4; ++j) {
        v16bf bb = load_frag16(&sB[(j * 16 + frow) * SBSTR + k0], fkh);
        acc[j] = __builtin_amdgcn_wmma_f32_16x16x32_bf16(false, a, false, bb,
                                                         (short)0, acc[j], false, false);
      }
    }
    // store this M tile; C/D layout: VGPR r -> row (lane>>4)*8+r, col lane&15
#pragma unroll
    for (int r = 0; r < 8; ++r) {
      int m = m0 + wave * 16 + mrow + r;
      float bbias = bias[m];
      size_t rowoff = (size_t)m * N + n0;
#pragma unroll
      for (int j = 0; j < 4; ++j) {
        float v = acc[j][r] + bbias;
        if (OUT_BF16) outb[rowoff + j * 16 + ncol] = (__bf16)v;
        else          outf[rowoff + j * 16 + ncol] = v;
      }
    }
  }
}

// ---------------------------------------------------------------------------
// Stage 2: window attention, one workgroup per (window, head).
// grid = (nWy*nWx, 6, B), block = 256 (8 wave32).
// off=0: aligned windows (V staged via global_load_async_to_lds_b128);
// off=-4: shifted windows (scalar staging, zero padding). Output bf16.
// ---------------------------------------------------------------------------
__global__ __launch_bounds__(256)
void win_attn(const __bf16* __restrict__ qkv,  // B x 1152 x HWPIX (bf16)
              __bf16* __restrict__ xout,       // B x 384 x HWPIX (bf16)
              int nWx, int off)
{
  __shared__ __attribute__((aligned(16))) __bf16 qs[64 * TSTR];  // [token][ch]
  __shared__ __attribute__((aligned(16))) __bf16 ks[64 * TSTR];  // [token][ch]
  __shared__ __attribute__((aligned(16))) __bf16 vt[64 * TSTR];  // [ch][token]
  __shared__ __attribute__((aligned(16))) float  sS[64 * SSTR];  // scores f32
  __shared__ __attribute__((aligned(16))) __bf16 ps[64 * TSTR];  // softmax bf16
  __shared__ float redMx[64 * 4];
  __shared__ float redSm[64 * 4];

  const int t    = threadIdx.x;
  const int lane = t & 31;
  const int wave = t >> 5;
  const int h    = blockIdx.y;
  const int b    = blockIdx.z;
  const int wy   = blockIdx.x / nWx;
  const int wx   = blockIdx.x % nWx;
  const int oy   = wy * 8 + off;
  const int ox   = wx * 8 + off;

  const __bf16* base  = qkv + (size_t)b * 1152 * HWPIX;
  const __bf16* vbase = base + (size_t)(768 + h * 64) * HWPIX;

  // ---- V staging first: async DMA per (channel, window-row) when the 8-token
  // row is in-bounds and 16B-aligned (always true for off==0). ----
  {
    const int ch = t & 63;
    const int r0 = t >> 6;                 // 0..3 -> rows r0, r0+4
    const __bf16* vrow = vbase + (size_t)ch * HWPIX;
#pragma unroll
    for (int rr = 0; rr < 2; ++rr) {
      int r = r0 + rr * 4;
      int y = oy + r;
      bool rowok = ((unsigned)y < (unsigned)IMG) && (ox >= 0) && (ox + 8 <= IMG);
      __bf16* dst = &vt[ch * TSTR + r * 8];
      if (rowok && ((off & 7) == 0)) {
        async_copy_b128((unsigned)(uintptr_t)dst, vrow + (size_t)y * IMG + ox);
      } else {
#pragma unroll
        for (int i = 0; i < 8; ++i) {
          int x = ox + i;
          bool ok = ((unsigned)y < (unsigned)IMG) && ((unsigned)x < (unsigned)IMG);
          dst[i] = ok ? vrow[(size_t)y * IMG + x] : (__bf16)0.0f;
        }
      }
    }
  }
  // ---- Q/K staging: bf16 copies with transpose to [token][ch] ----
  {
    const int ch   = t >> 2;
    const int tok0 = (t & 3) * 16;
    const __bf16* qrow = base + (size_t)(  0 + h * 64 + ch) * HWPIX;
    const __bf16* krow = base + (size_t)(384 + h * 64 + ch) * HWPIX;
#pragma unroll
    for (int i = 0; i < 16; ++i) {
      int n = tok0 + i;
      int y = oy + (n >> 3);
      int x = ox + (n & 7);
      bool ok = ((unsigned)y < (unsigned)IMG) && ((unsigned)x < (unsigned)IMG);
      int p = y * IMG + x;
      qs[n * TSTR + ch] = ok ? qrow[p] : (__bf16)0.0f;
      ks[n * TSTR + ch] = ok ? krow[p] : (__bf16)0.0f;
    }
  }
  wait_async0();      // this wave's async V copies landed in LDS
  __syncthreads();    // everyone's copies visible

  const int frow = lane & 15;
  const int fkh  = (lane >> 4) * 8;
  const int mrow = (lane >> 4) * 8;
  const int ncol = lane & 15;

  // ---- S = (Q @ K^T) / sqrt(64): 16 tiles of 16x16, 2 per wave ----
#pragma unroll
  for (int tt = 0; tt < 2; ++tt) {
    int tile = wave * 2 + tt;
    int mi = tile >> 2, ni = tile & 3;
    v8f acc = {};
#pragma unroll
    for (int kk = 0; kk < 64; kk += 32) {
      v16bf a  = load_frag16(&qs[(mi * 16 + frow) * TSTR + kk], fkh);
      v16bf bb = load_frag16(&ks[(ni * 16 + frow) * TSTR + kk], fkh);
      acc = __builtin_amdgcn_wmma_f32_16x16x32_bf16(false, a, false, bb,
                                                    (short)0, acc, false, false);
    }
#pragma unroll
    for (int r = 0; r < 8; ++r)
      sS[(mi * 16 + mrow + r) * SSTR + ni * 16 + ncol] = acc[r] * 0.125f;
  }
  __syncthreads();

  // ---- row softmax, 4 threads per row (row = t&63, quarter = t>>6) ----
  {
    const int row = t & 63;
    const int q   = t >> 6;
    const float* rp = &sS[row * SSTR + q * 16];
    float mx = -3.402823466e38f;
#pragma unroll
    for (int i = 0; i < 16; ++i) mx = fmaxf(mx, rp[i]);
    redMx[row * 4 + q] = mx;
    __syncthreads();
    mx = fmaxf(fmaxf(redMx[row * 4 + 0], redMx[row * 4 + 1]),
               fmaxf(redMx[row * 4 + 2], redMx[row * 4 + 3]));
    float e[16];
    float sum = 0.0f;
#pragma unroll
    for (int i = 0; i < 16; ++i) { e[i] = __expf(rp[i] - mx); sum += e[i]; }
    redSm[row * 4 + q] = sum;
    __syncthreads();
    sum = (redSm[row * 4 + 0] + redSm[row * 4 + 1]) +
          (redSm[row * 4 + 2] + redSm[row * 4 + 3]);
    float inv = __frcp_rn(sum);
#pragma unroll
    for (int i = 0; i < 16; ++i)
      ps[row * TSTR + q * 16 + i] = (__bf16)(e[i] * inv);
  }
  __syncthreads();

  // ---- O = P @ V, scatter bf16 to xout ----
  __bf16* outB = xout + (size_t)b * 384 * HWPIX;
#pragma unroll
  for (int tt = 0; tt < 2; ++tt) {
    int tile = wave * 2 + tt;
    int mi = tile >> 2, ni = tile & 3;
    v8f acc = {};
#pragma unroll
    for (int kk = 0; kk < 64; kk += 32) {
      v16bf a  = load_frag16(&ps[(mi * 16 + frow) * TSTR + kk], fkh);
      v16bf bb = load_frag16(&vt[(ni * 16 + frow) * TSTR + kk], fkh);
      acc = __builtin_amdgcn_wmma_f32_16x16x32_bf16(false, a, false, bb,
                                                    (short)0, acc, false, false);
    }
#pragma unroll
    for (int r = 0; r < 8; ++r) {
      int m = mi * 16 + mrow + r;   // token
      int n = ni * 16 + ncol;       // channel within head
      int y = oy + (m >> 3);
      int x = ox + (m & 7);
      if (((unsigned)y < (unsigned)IMG) && ((unsigned)x < (unsigned)IMG))
        outB[(size_t)(h * 64 + n) * HWPIX + (size_t)y * IMG + x] = (__bf16)acc[r];
    }
  }
}

// ---------------------------------------------------------------------------
extern "C" void kernel_launch(void* const* d_in, const int* in_sizes, int n_in,
                              void* d_out, int out_size, void* d_ws, size_t ws_size,
                              hipStream_t stream) {
  (void)in_sizes; (void)n_in; (void)out_size; (void)ws_size;
  const float* x      = (const float*)d_in[0];  // 4 x 384 x 128 x 128
  const float* w_qkv  = (const float*)d_in[1];  // 1152 x 384
  const float* b_qkv  = (const float*)d_in[2];  // 1152
  const float* w_head = (const float*)d_in[3];  // 384 x 384
  const float* b_head = (const float*)d_in[4];  // 384
  float* out = (float*)d_out;                   // 4 x 384 x 128 x 128

  __bf16* qkv = (__bf16*)d_ws;                         // 4*1152*16384 bf16 (144 MB)
  __bf16* x1  = qkv + (size_t)4 * 1152 * HWPIX;        // 4* 384*16384 bf16 ( 48 MB)
  __bf16* x2  = x1  + (size_t)4 *  384 * HWPIX;        // 4* 384*16384 bf16 ( 48 MB)

  // Stage 1: qkv projection (f32 act -> bf16 out)
  wmma_gemm_full<false, true><<<dim3(HWPIX / 64, 1, 4), 256, 0, stream>>>(
      w_qkv, x, nullptr, b_qkv, qkv, 1152, HWPIX);

  // Stage 2: aligned windows (async V staging), then shifted windows
  win_attn<<<dim3(16 * 16, 6, 4), 256, 0, stream>>>(qkv, x1, 16,  0);
  win_attn<<<dim3(17 * 17, 6, 4), 256, 0, stream>>>(qkv, x2, 17, -4);

  // Stage 3: head projection on (x1 + x2) (bf16 acts -> f32 out)
  wmma_gemm_full<true, false><<<dim3(HWPIX / 64, 1, 4), 256, 0, stream>>>(
      w_head, x1, x2, b_head, out, 384, HWPIX);
}
